// CrossAttention_1683627180152
// MI455X (gfx1250) — matrix operand order repaired and checked by reading, not compiled, not run on hardware
//
#include <hip/hip_runtime.h>

typedef __attribute__((ext_vector_type(16))) _Float16 v16h;
typedef __attribute__((ext_vector_type(8)))  _Float16 v8h;
typedef __attribute__((ext_vector_type(8)))  float    v8f;

union u16h { v16h v; v8h h[2]; };

#define BATCH  2
#define NQ     4096
#define MKEY   4096
#define DIMC   512
#define NHEAD  8
#define HD     64
#define MPAD   4160                       // 4097 keys (null + 4096) padded to x64
#define NTILES (MPAD / 64)
#define SSCALE 0.3535533905932738f        // 64^(-0.25), applied to both q and k

// ---------------------------------------------------------------- utilities

__global__ void k_cast16(const float* __restrict__ in, _Float16* __restrict__ out, int n) {
    int i = blockIdx.x * blockDim.x + threadIdx.x;
    if (i < n) out[i] = (_Float16)in[i];
}

// Wt[n][k] = W[k][n], f32 -> f16 (weights stored K-major so B fragments are contiguous)
__global__ void k_transpose_cast(const float* __restrict__ W, _Float16* __restrict__ Wt,
                                 int K, int N) {
    int i = blockIdx.x * blockDim.x + threadIdx.x;
    if (i < K * N) {
        int k = i / N, n = i % N;
        Wt[(size_t)n * K + k] = (_Float16)W[i];
    }
}

// Row layernorm over 512 cols; 256 threads, 2 elems/thread.
template <typename OT>
__global__ void k_layernorm(const float* __restrict__ x, const float* __restrict__ gw,
                            OT* __restrict__ out) {
    int row = blockIdx.x;
    int t = threadIdx.x;
    const float* xr = x + (size_t)row * 512;
    float a = xr[t], b = xr[t + 256];
    float s = a + b;
    float ss = a * a + b * b;
    for (int m = 1; m < 32; m <<= 1) {
        s  += __shfl_xor(s,  m, 32);
        ss += __shfl_xor(ss, m, 32);
    }
    __shared__ float red[16];
    int wv = t >> 5;
    if ((t & 31) == 0) { red[wv] = s; red[wv + 8] = ss; }
    __syncthreads();
    float ts = 0.f, tss = 0.f;
    for (int i = 0; i < 8; i++) { ts += red[i]; tss += red[i + 8]; }
    float mean = ts * (1.0f / 512.0f);
    float var  = tss * (1.0f / 512.0f) - mean * mean;
    float rstd = rsqrtf(var + 1e-5f);
    out[(size_t)row * 512 + t]       = (OT)((a - mean) * rstd * gw[t]);
    out[(size_t)row * 512 + t + 256] = (OT)((b - mean) * rstd * gw[t + 256]);
}

// null kv at key index 0, zero padding rows [4097, MPAD)
__global__ void k_nullpad(const float* __restrict__ null_kv,
                          _Float16* __restrict__ k16, _Float16* __restrict__ vT) {
    int t = blockIdx.x * blockDim.x + threadIdx.x;
    if (t >= BATCH * NHEAD * HD) return;
    int bh = t >> 6, d = t & 63;
    k16[((size_t)bh * MPAD) * 64 + d]   = (_Float16)(null_kv[d] * SSCALE);
    vT[((size_t)bh * 64 + d) * MPAD]    = (_Float16)null_kv[64 + d];
    for (int j = MKEY + 1; j < MPAD; j++) {
        k16[((size_t)bh * MPAD + j) * 64 + d] = (_Float16)0.f;
        vT[((size_t)bh * 64 + d) * MPAD + j]  = (_Float16)0.f;
    }
}

// per-key additive logit bias: 0 for null key & unmasked, -1e30 for masked/pad
__global__ void k_bias(const int* __restrict__ mask, float* __restrict__ bias) {
    int i = blockIdx.x * blockDim.x + threadIdx.x;
    if (i >= BATCH * MPAD) return;
    int b = i / MPAD, j = i % MPAD;
    float v;
    if (j == 0) v = 0.f;
    else if (j <= MKEY) v = mask[b * MKEY + (j - 1)] ? 0.f : -1.0e30f;
    else v = -1.0e30f;
    bias[i] = v;
}

// ------------------------------------------------- generic WMMA GEMM, 16x64/wave
// A: [M][512] f16 row-major.  Bt: [Ncols][512] f16 (i.e. B transposed).
// mode 0: q projection   -> out0 = q16  [b][h][n][64]   (scaled)
// mode 1: kv projection  -> out0 = k16  [b][h][MPAD][64] (scaled, at j+1)
//                           out1 = vT   [b][h][64][MPAD] (at j+1)
// mode 2: o projection   -> outf = f32  [M][512] row-major
__global__ void __launch_bounds__(128)
k_gemm(const _Float16* __restrict__ A, const _Float16* __restrict__ Bt,
       int tilesN, int mode, float scale,
       _Float16* __restrict__ out0, _Float16* __restrict__ out1, float* __restrict__ outf) {
    int wid  = (blockIdx.x * blockDim.x + threadIdx.x) >> 5;
    int lane = threadIdx.x & 31;
    int g = lane >> 4, ln = lane & 15;
    int m0 = (wid / tilesN) * 16;
    int n0 = (wid % tilesN) * 64;

    v8f c[4] = {};
    const _Float16* arow = A + (size_t)(m0 + ln) * 512;
    for (int k0 = 0; k0 < 512; k0 += 32) {
        u16h a;
        a.h[0] = *(const v8h*)(arow + k0 + 16 * g);
        a.h[1] = *(const v8h*)(arow + k0 + 16 * g + 8);
        for (int nt = 0; nt < 4; nt++) {
            const _Float16* bp = Bt + (size_t)(n0 + nt * 16 + ln) * 512 + k0 + 16 * g;
            v16h b = *(const v16h*)bp;
            c[nt] = __builtin_amdgcn_wmma_f32_16x16x32_f16(false, a.v, false, b,
                                                           (short)0, c[nt], false, false);
        }
    }

    for (int nt = 0; nt < 4; nt++)
        for (int v = 0; v < 8; v++) {
            float val = c[nt][v];
            int gr = m0 + v + 8 * g;
            int gc = n0 + nt * 16 + ln;
            if (mode == 0) {
                int b = gr >> 12, i = gr & 4095;
                int h = gc >> 6,  d = gc & 63;
                out0[((size_t)(b * NHEAD + h) * NQ + i) * 64 + d] = (_Float16)(val * scale);
            } else if (mode == 1) {
                int b = gr >> 12, j = gr & 4095;
                if (gc < 512) {
                    int h = gc >> 6, d = gc & 63;
                    out0[((size_t)(b * NHEAD + h) * MPAD + 1 + j) * 64 + d] =
                        (_Float16)(val * scale);
                } else {
                    int cc = gc - 512;
                    int h = cc >> 6, d = cc & 63;
                    out1[((size_t)(b * NHEAD + h) * 64 + d) * MPAD + 1 + j] = (_Float16)val;
                }
            } else {
                outf[(size_t)gr * 512 + gc] = val;
            }
        }
}

// ------------------------------------------------------- flash attention core
// Block = 4 waves, all on the same (b,h): 64 query rows. K/V tiles (64 keys)
// are staged cooperatively into double-buffered LDS with async-to-LDS copies
// (ASYNCcnt) so the copy of tile j+1 overlaps the WMMA work on tile j.
__global__ void __launch_bounds__(128)
k_flash(const _Float16* __restrict__ q16, const _Float16* __restrict__ k16,
        const _Float16* __restrict__ vT, const float* __restrict__ bias,
        _Float16* __restrict__ attn16) {
    __shared__ __align__(32) _Float16 kbuf[2][64 * 64];   // [key][d], 8KB each
    __shared__ __align__(32) _Float16 vbuf[2][64 * 64];   // [d][key], 8KB each
    __shared__ __align__(32) _Float16 plds_all[4 * 16 * 64];

    int tid  = threadIdx.x;
    int wv   = tid >> 5;
    int lane = tid & 31;
    int g = lane >> 4, ln = lane & 15;
    int bh = (blockIdx.x * 4) >> 8;            // uniform across the block
    int gid = blockIdx.x * 4 + wv;
    int i0 = (gid & 255) << 4;
    const _Float16* qb = q16 + ((size_t)bh * NQ + i0) * 64;
    const _Float16* kb = k16 + (size_t)bh * MPAD * 64;    // uniform -> SGPR base
    const _Float16* vb = vT  + (size_t)bh * 64 * MPAD;    // uniform -> SGPR base
    const float* biasb = bias + (size_t)(bh >> 3) * MPAD;
    _Float16* pl = plds_all + wv * 1024;

    // Cooperative async copy of one 64-key tile (K: 8KB contiguous; V: 64 rows
    // of 128B from the transposed global V). 8 x b128 per thread.
    auto issue = [&](int buf, int j0) {
        unsigned kbase = (unsigned)(uintptr_t)(&kbuf[buf][0]);
        unsigned vbase = (unsigned)(uintptr_t)(&vbuf[buf][0]);
        for (int i = 0; i < 4; i++) {
            int idx = tid + 128 * i;               // 0..511
            unsigned l = kbase + (unsigned)(idx * 16);
            unsigned o = (unsigned)(j0 * 128 + idx * 16);
            asm volatile("global_load_async_to_lds_b128 %0, %1, %2 offset:0"
                         :: "v"(l), "v"(o), "s"(kb) : "memory");
        }
        for (int i = 0; i < 4; i++) {
            int idx = tid + 128 * i;
            int d = idx >> 3, seg = idx & 7;
            unsigned l = vbase + (unsigned)((d * 64 + seg * 8) * 2);
            unsigned o = (unsigned)((d * MPAD + j0 + seg * 8) * 2);
            asm volatile("global_load_async_to_lds_b128 %0, %1, %2 offset:0"
                         :: "v"(l), "v"(o), "s"(vb) : "memory");
        }
    };

    // resident Q fragments (16 x 64 over two K=32 chunks)
    u16h qa[2];
    for (int c = 0; c < 2; c++) {
        const _Float16* qr = qb + ln * 64 + 32 * c + 16 * g;
        qa[c].h[0] = *(const v8h*)qr;
        qa[c].h[1] = *(const v8h*)(qr + 8);
    }

    v8f O[4] = {};
    float mrow[8], lrow[8];
    for (int v = 0; v < 8; v++) { mrow[v] = -3.0e38f; lrow[v] = 0.f; }

    issue(0, 0);   // prefetch first tile

    for (int jt = 0; jt < NTILES; jt++) {
        int j0 = jt * 64;
        int buf = jt & 1;

        asm volatile("s_wait_asynccnt 0x0" ::: "memory");  // my copies for `buf` done
        __syncthreads();                                   // everyone's copies done,
                                                           // prev reads of buf^1 done
        if (jt + 1 < NTILES) issue(buf ^ 1, j0 + 64);

        float bl[4];
        for (int nt = 0; nt < 4; nt++) bl[nt] = biasb[j0 + nt * 16 + ln];

        // S = Q K^T  (B fragments from the LDS K tile)
        v8f S[4];
        for (int nt = 0; nt < 4; nt++) {
            const _Float16* kr = &kbuf[buf][(nt * 16 + ln) * 64];
            v16h b0 = *(const v16h*)(kr + 16 * g);
            v16h b1 = *(const v16h*)(kr + 32 + 16 * g);
            v8f acc = {};
            acc = __builtin_amdgcn_wmma_f32_16x16x32_f16(false, qa[0].v, false, b0,
                                                         (short)0, acc, false, false);
            acc = __builtin_amdgcn_wmma_f32_16x16x32_f16(false, qa[1].v, false, b1,
                                                         (short)0, acc, false, false);
            S[nt] = acc;
        }
        for (int nt = 0; nt < 4; nt++)
            for (int v = 0; v < 8; v++) S[nt][v] += bl[nt];

        // online softmax: row max over this tile (reduce across 16-lane group)
        float mt[8];
        for (int v = 0; v < 8; v++) {
            float mx = fmaxf(fmaxf(S[0][v], S[1][v]), fmaxf(S[2][v], S[3][v]));
            mx = fmaxf(mx, __shfl_xor(mx, 1, 32));
            mx = fmaxf(mx, __shfl_xor(mx, 2, 32));
            mx = fmaxf(mx, __shfl_xor(mx, 4, 32));
            mx = fmaxf(mx, __shfl_xor(mx, 8, 32));
            mt[v] = mx;
        }
        float alpha[8];
        for (int v = 0; v < 8; v++) {
            float mn = fmaxf(mrow[v], mt[v]);
            alpha[v] = __expf(mrow[v] - mn);
            mrow[v] = mn;
        }
        float rs[8];
        for (int v = 0; v < 8; v++) rs[v] = 0.f;
        for (int nt = 0; nt < 4; nt++)
            for (int v = 0; v < 8; v++) {
                float p = __expf(S[nt][v] - mrow[v]);
                S[nt][v] = p;
                rs[v] += p;
            }
        for (int v = 0; v < 8; v++) {
            float s = rs[v];
            s += __shfl_xor(s, 1, 32);
            s += __shfl_xor(s, 2, 32);
            s += __shfl_xor(s, 4, 32);
            s += __shfl_xor(s, 8, 32);
            lrow[v] = lrow[v] * alpha[v] + s;
        }
        for (int dt = 0; dt < 4; dt++)
            for (int v = 0; v < 8; v++) O[dt][v] *= alpha[v];

        // re-stripe P (C-layout -> A-layout) through the per-wave LDS tile
        for (int nt = 0; nt < 4; nt++)
            for (int v = 0; v < 8; v++)
                pl[(v + 8 * g) * 64 + nt * 16 + ln] = (_Float16)S[nt][v];
        u16h pa[2];
        for (int c = 0; c < 2; c++) {
            const _Float16* pr = pl + ln * 64 + 32 * c + 16 * g;
            pa[c].h[0] = *(const v8h*)pr;
            pa[c].h[1] = *(const v8h*)(pr + 8);
        }

        // O += P V   (B fragments from the LDS V tile, [d][key] layout)
        for (int dt = 0; dt < 4; dt++) {
            v8f acc = O[dt];
            for (int c = 0; c < 2; c++) {
                const _Float16* vr = &vbuf[buf][(dt * 16 + ln) * 64 + 32 * c + 16 * g];
                v16h bv = *(const v16h*)vr;
                acc = __builtin_amdgcn_wmma_f32_16x16x32_f16(false, pa[c].v, false, bv,
                                                             (short)0, acc, false, false);
            }
            O[dt] = acc;
        }
    }

    // finalize: divide by row sums, write [b*n][512] f16 for the Wo GEMM
    int b = bh >> 3, h = bh & 7;
    for (int v = 0; v < 8; v++) {
        float inv = 1.0f / lrow[v];
        size_t row = (size_t)b * NQ + i0 + v + 8 * g;
        for (int dt = 0; dt < 4; dt++)
            attn16[row * 512 + h * 64 + dt * 16 + ln] = (_Float16)(O[dt][v] * inv);
    }
}

// ---------------------------------------------------------------- launcher

extern "C" void kernel_launch(void* const* d_in, const int* in_sizes, int n_in,
                              void* d_out, int out_size, void* d_ws, size_t ws_size,
                              hipStream_t stream) {
    const float* x       = (const float*)d_in[0];
    const float* context = (const float*)d_in[1];
    const int*   mask    = (const int*)d_in[2];
    const float* g_x     = (const float*)d_in[3];
    const float* null_kv = (const float*)d_in[4];
    const float* Wq      = (const float*)d_in[5];
    const float* Wkv     = (const float*)d_in[6];
    const float* Wo      = (const float*)d_in[7];
    const float* g_out   = (const float*)d_in[8];
    float* outp = (float*)d_out;
    char* ws = (char*)d_ws;

    _Float16* xn16  = (_Float16*)(ws + 0);          // 8,388,608 B ; reused as attn16
    _Float16* ctx16 = (_Float16*)(ws + 8388608);    // 8,388,608 B
    _Float16* WqT   = (_Float16*)(ws + 16777216);   //   524,288 B
    _Float16* WkvT  = (_Float16*)(ws + 17301504);   // 1,048,576 B
    _Float16* WoT   = (_Float16*)(ws + 18350080);   //   524,288 B
    _Float16* q16   = (_Float16*)(ws + 18874368);   // 8,388,608 B
    _Float16* k16   = (_Float16*)(ws + 27262976);   // 8,519,680 B
    _Float16* vT    = (_Float16*)(ws + 35782656);   // 8,519,680 B
    float*    bias  = (float*)   (ws + 44302336);   //    33,280 B
    float*    obuf  = (float*)   (ws + 18874368);   // 16,777,216 B, aliases q16/k16 (dead)
    _Float16* attn16 = xn16;                        // aliases xn16 (dead after q GEMM)

    k_transpose_cast<<<(512 * 512 + 255) / 256, 256, 0, stream>>>(Wq,  WqT,  512, 512);
    k_transpose_cast<<<(512 * 1024 + 255) / 256, 256, 0, stream>>>(Wkv, WkvT, 512, 1024);
    k_transpose_cast<<<(512 * 512 + 255) / 256, 256, 0, stream>>>(Wo,  WoT,  512, 512);

    k_layernorm<_Float16><<<BATCH * NQ, 256, 0, stream>>>(x, g_x, xn16);
    k_cast16<<<(BATCH * MKEY * 512 + 255) / 256, 256, 0, stream>>>(context, ctx16,
                                                                   BATCH * MKEY * 512);
    k_nullpad<<<4, 256, 0, stream>>>(null_kv, k16, vT);
    k_bias<<<(BATCH * MPAD + 255) / 256, 256, 0, stream>>>(mask, bias);

    // Q = LN(x) @ Wq : M=8192, Ncols=512 -> 4096 waves
    k_gemm<<<1024, 128, 0, stream>>>(xn16, WqT, 8, 0, SSCALE, q16, nullptr, nullptr);
    // KV = ctx @ Wkv : M=8192, Ncols=1024 -> 8192 waves
    k_gemm<<<2048, 128, 0, stream>>>(ctx16, WkvT, 16, 1, SSCALE, k16, vT, nullptr);

    // flash attention: 2*8*4096/16 = 4096 waves
    k_flash<<<1024, 128, 0, stream>>>(q16, k16, vT, bias, attn16);

    // O = attn @ Wo -> f32, then layernorm
    k_gemm<<<1024, 128, 0, stream>>>(attn16, WoT, 8, 2, 1.0f, nullptr, nullptr, obuf);
    k_layernorm<float><<<BATCH * NQ, 256, 0, stream>>>(obuf, g_out, outp);
}
